// AttentionDecoder_43834436223362
// MI455X (gfx1250) — matrix operand order repaired and checked by reading, not compiled, not run on hardware
//
#include <hip/hip_runtime.h>

// ---------------------------------------------------------------------------
// Problem constants (AttentionDecoder reference)
// ---------------------------------------------------------------------------
#define BB   128      // batch
#define SS   196      // encoder seq len
#define KK   512      // keys hidden size
#define TT   15       // output steps (T_CAP - 1)
#define TCAP 16
#define VV   32000    // vocab
#define HH   512      // hidden / embedding

typedef __attribute__((ext_vector_type(8)))  __bf16 v8bf;
typedef __attribute__((ext_vector_type(16))) __bf16 v16bf;
typedef __attribute__((ext_vector_type(8)))  float  v8f;

// ---------------------------------------------------------------------------
// f32 -> bf16 conversion
// ---------------------------------------------------------------------------
__global__ __launch_bounds__(256) void cvt_f32_bf16(const float* __restrict__ src,
                                                    __bf16* __restrict__ dst, int n) {
  int i = blockIdx.x * 256 + threadIdx.x;
  if (i < n) dst[i] = (__bf16)src[i];
}

// gather embedding rows for captions[:, :TT] -> xg (B*TT, H) bf16
__global__ __launch_bounds__(256) void gather_embed(const float* __restrict__ emb,
                                                    const int* __restrict__ cap,
                                                    __bf16* __restrict__ xg) {
  int i = blockIdx.x * 256 + threadIdx.x;
  if (i >= BB * TT * HH) return;
  int e   = i & (HH - 1);
  int row = i >> 9;             // HH == 512
  int b   = row / TT;
  int t   = row - b * TT;
  int tok = cap[b * TCAP + t];
  xg[i] = (__bf16)emb[(long)tok * HH + e];
}

__global__ __launch_bounds__(256) void init_h(float* __restrict__ hf,
                                              __bf16* __restrict__ hb) {
  int i = blockIdx.x * 256 + threadIdx.x;
  if (i < BB * HH) { hf[i] = 0.0f; hb[i] = (__bf16)0.0f; }
}

// ---------------------------------------------------------------------------
// Generic bf16 WMMA GEMM:  C[M,N] = sum_k A0[m,k]*W0[n,k] (+ A1@W1^T) (+ bias)
// Wave: one 16x16 M-tile x four consecutive N-tiles (A-fragment reuse x4).
// Block: 8 waves -> 32 N-tiles per block.
// ---------------------------------------------------------------------------
static __device__ __forceinline__ void gemm_pass(const __bf16* __restrict__ A, int lda,
                                                 const __bf16* __restrict__ W, int ldw, int K,
                                                 int mrow, int n0, int n1, int n2, int n3,
                                                 int lr, int half, v8f acc[4]) {
  const __bf16* ap  = A + (long)mrow * lda + half * 16;
  const __bf16* bp0 = W + (long)(n0 * 16 + lr) * ldw + half * 16;
  const __bf16* bp1 = W + (long)(n1 * 16 + lr) * ldw + half * 16;
  const __bf16* bp2 = W + (long)(n2 * 16 + lr) * ldw + half * 16;
  const __bf16* bp3 = W + (long)(n3 * 16 + lr) * ldw + half * 16;
  for (int k0 = 0; k0 < K; k0 += 32) {
    // A frag: lane holds row M=lr, K = half*16 + [0..15]
    v8bf alo = *(const v8bf*)(ap);
    v8bf ahi = *(const v8bf*)(ap + 8);
    v16bf a = __builtin_shufflevector(alo, ahi,
                                      0, 1, 2, 3, 4, 5, 6, 7,
                                      8, 9, 10, 11, 12, 13, 14, 15);
    // B frag: lane holds col N=lr, K = half*16 + [0..15] (contiguous weight row)
    v16bf b0 = *(const v16bf*)bp0;
    v16bf b1 = *(const v16bf*)bp1;
    v16bf b2 = *(const v16bf*)bp2;
    v16bf b3 = *(const v16bf*)bp3;
    acc[0] = __builtin_amdgcn_wmma_f32_16x16x32_bf16(false, a, false, b0, (short)0, acc[0], false, false);
    acc[1] = __builtin_amdgcn_wmma_f32_16x16x32_bf16(false, a, false, b1, (short)0, acc[1], false, false);
    acc[2] = __builtin_amdgcn_wmma_f32_16x16x32_bf16(false, a, false, b2, (short)0, acc[2], false, false);
    acc[3] = __builtin_amdgcn_wmma_f32_16x16x32_bf16(false, a, false, b3, (short)0, acc[3], false, false);
    ap += 32; bp0 += 32; bp1 += 32; bp2 += 32; bp3 += 32;
  }
}

__global__ __launch_bounds__(256) void gemm_bf16_kernel(
    const __bf16* __restrict__ A0, int lda0, const __bf16* __restrict__ W0, int ldw0, int K0,
    const __bf16* __restrict__ A1, int lda1, const __bf16* __restrict__ W1, int ldw1, int K1,
    const float* __restrict__ bias,
    float* __restrict__ C, long ldc,
    __bf16* __restrict__ Cb, long ldcb,
    int Ntiles) {
  const int wave  = threadIdx.x >> 5;
  const int lane  = threadIdx.x & 31;
  const int half  = lane >> 4;
  const int lr    = lane & 15;
  const int mtile = blockIdx.y;
  const int nbase = blockIdx.x * 32 + wave * 4;
  if (nbase >= Ntiles) return;                 // whole-wave exit: WMMA waves keep EXEC all-ones
  int n0 = nbase,     n1 = nbase + 1, n2 = nbase + 2, n3 = nbase + 3;
  if (n1 >= Ntiles) n1 = Ntiles - 1;           // clamp (redundant tiles, stores guarded)
  if (n2 >= Ntiles) n2 = Ntiles - 1;
  if (n3 >= Ntiles) n3 = Ntiles - 1;

  const int mrow = mtile * 16 + lr;
  v8f z = {0.f, 0.f, 0.f, 0.f, 0.f, 0.f, 0.f, 0.f};
  v8f acc[4] = {z, z, z, z};

  gemm_pass(A0, lda0, W0, ldw0, K0, mrow, n0, n1, n2, n3, lr, half, acc);
  if (A1) gemm_pass(A1, lda1, W1, ldw1, K1, mrow, n0, n1, n2, n3, lr, half, acc);

  int nts[4] = {n0, n1, n2, n3};
#pragma unroll
  for (int j = 0; j < 4; ++j) {
    if (nbase + j >= Ntiles) break;
    const int ncol = nts[j] * 16 + lr;
    const float bv = bias ? bias[ncol] : 0.0f;
#pragma unroll
    for (int r = 0; r < 8; ++r) {
      // D layout: VGPR r -> M = mtile*16 + half*8 + r, N = ncol
      const int mr = mtile * 16 + half * 8 + r;
      const float v = acc[j][r] + bv;
      if (C)  C[(long)mr * ldc + ncol] = v;
      if (Cb) Cb[(long)mr * ldcb + ncol] = (__bf16)v;
    }
  }
}

// ---------------------------------------------------------------------------
// Additive attention + softmax + context (one block per batch row)
// scores[s] = tanh(q[b] + keys_proj[b,s]) . att_v ; w = softmax(scores)
// ctx[b,k]  = sum_s w[s] * features[b,s,k]         (written as bf16 for fc0)
// ---------------------------------------------------------------------------
__global__ __launch_bounds__(256) void attention_kernel(
    const float* __restrict__ q,      // (B,H)
    const float* __restrict__ keysp,  // (B,S,H)
    const float* __restrict__ attv,   // (H)
    const float* __restrict__ feat,   // (B,S,K)
    __bf16* __restrict__ ctx_bf) {    // (B,K)
  __shared__ float s_q[HH];
  __shared__ float s_v[HH];
  __shared__ float s_w[SS];
  __shared__ float redm[8];
  __shared__ float reds[8];

  const int b = blockIdx.x, tid = threadIdx.x;
  const int wave = tid >> 5, lane = tid & 31;

  for (int i = tid; i < HH; i += 256) { s_q[i] = q[b * HH + i]; s_v[i] = attv[i]; }
  __syncthreads();

  // scores: one wave per encoder position
  for (int s = wave; s < SS; s += 8) {
    const float* kp = keysp + ((long)b * SS + s) * HH;
    float acc = 0.0f;
    for (int h = lane; h < HH; h += 32) acc += tanhf(s_q[h] + kp[h]) * s_v[h];
    for (int off = 16; off; off >>= 1) acc += __shfl_xor(acc, off, 32);
    if (lane == 0) s_w[s] = acc;
  }
  __syncthreads();

  // softmax over S (each score owned by exactly one thread: s == tid < SS)
  float lm = -3.402823e38f;
  for (int s = tid; s < SS; s += 256) lm = fmaxf(lm, s_w[s]);
  for (int off = 16; off; off >>= 1) lm = fmaxf(lm, __shfl_xor(lm, off, 32));
  if (lane == 0) redm[wave] = lm;
  __syncthreads();
  float gmax = redm[0];
  for (int i = 1; i < 8; ++i) gmax = fmaxf(gmax, redm[i]);

  float ls = 0.0f;
  for (int s = tid; s < SS; s += 256) ls += __expf(s_w[s] - gmax);
  for (int off = 16; off; off >>= 1) ls += __shfl_xor(ls, off, 32);
  if (lane == 0) reds[wave] = ls;
  __syncthreads();
  float gsum = 0.0f;
  for (int i = 0; i < 8; ++i) gsum += reds[i];
  const float inv = 1.0f / gsum;

  for (int s = tid; s < SS; s += 256) s_w[s] = __expf(s_w[s] - gmax) * inv;
  __syncthreads();

  // context = w . features  (features stay L2-resident across steps)
  for (int k = tid; k < KK; k += 256) {
    const float* fb = feat + (long)b * SS * KK + k;
    float acc = 0.0f;
    for (int s = 0; s < SS; ++s) acc += s_w[s] * fb[(long)s * KK];
    ctx_bf[b * KK + k] = (__bf16)acc;
  }
}

// ---------------------------------------------------------------------------
// GRU gates (torch order r,z,n); updates h in place (f32 + bf16 copy)
// ---------------------------------------------------------------------------
__global__ __launch_bounds__(256) void gru_gates_kernel(
    const float* __restrict__ gi,   // (B, 3H) = x/ctx part + bih
    const float* __restrict__ gh,   // (B, 3H) = h part + bhh
    float* __restrict__ hf,         // (B, H) in/out
    __bf16* __restrict__ hb) {      // (B, H) out
  int i = blockIdx.x * 256 + threadIdx.x;
  if (i >= BB * HH) return;
  int b = i >> 9, hcol = i & (HH - 1);
  long base = (long)b * (3 * HH);
  float ir = gi[base + hcol],          hr = gh[base + hcol];
  float iz = gi[base + HH + hcol],     hz = gh[base + HH + hcol];
  float in = gi[base + 2 * HH + hcol], hn = gh[base + 2 * HH + hcol];
  float r = 1.0f / (1.0f + __expf(-(ir + hr)));
  float z = 1.0f / (1.0f + __expf(-(iz + hz)));
  float n = tanhf(in + r * hn);
  float hprev = hf[i];
  float hnew = (1.0f - z) * n + z * hprev;
  hf[i] = hnew;
  hb[i] = (__bf16)hnew;
}

// ---------------------------------------------------------------------------
// Host launcher
// ---------------------------------------------------------------------------
extern "C" void kernel_launch(void* const* d_in, const int* in_sizes, int n_in,
                              void* d_out, int out_size, void* d_ws, size_t ws_size,
                              hipStream_t stream) {
  const float* features = (const float*)d_in[0];
  const int*   captions = (const int*)d_in[1];
  const float* emb      = (const float*)d_in[2];
  const float* fc1_w    = (const float*)d_in[3];
  const float* fc1_b    = (const float*)d_in[4];
  const float* att_wq   = (const float*)d_in[5];
  const float* att_wk   = (const float*)d_in[6];
  const float* att_v    = (const float*)d_in[7];
  const float* fc0_w    = (const float*)d_in[8];
  const float* fc0_b    = (const float*)d_in[9];
  const float* gru_wih  = (const float*)d_in[10];
  const float* gru_whh  = (const float*)d_in[11];
  const float* gru_bih  = (const float*)d_in[12];
  const float* gru_bhh  = (const float*)d_in[13];
  const float* fc2_w    = (const float*)d_in[14];
  const float* fc2_b    = (const float*)d_in[15];
  float* out = (float*)d_out;

  // ---- workspace layout (256B aligned blocks) ----
  char* ws = (char*)d_ws;
  size_t off = 0;
  auto alloc = [&](size_t bytes) {
    void* p = (void*)(ws + off);
    off += (bytes + 255) & ~(size_t)255;
    return p;
  };
  __bf16* fc1w_b  = (__bf16*)alloc((size_t)HH * HH * 2);
  __bf16* attwq_b = (__bf16*)alloc((size_t)HH * HH * 2);
  __bf16* attwk_b = (__bf16*)alloc((size_t)HH * KK * 2);
  __bf16* fc0w_b  = (__bf16*)alloc((size_t)HH * KK * 2);
  __bf16* wih_b   = (__bf16*)alloc((size_t)3 * HH * 2 * HH * 2);
  __bf16* whh_b   = (__bf16*)alloc((size_t)3 * HH * HH * 2);
  __bf16* fc2w_b  = (__bf16*)alloc((size_t)VV * HH * 2);
  __bf16* feat_b  = (__bf16*)alloc((size_t)BB * SS * KK * 2);
  __bf16* xg_b    = (__bf16*)alloc((size_t)BB * TT * HH * 2);
  __bf16* xw_b    = (__bf16*)alloc((size_t)BB * TT * HH * 2);
  float*  keysp   = (float*)alloc((size_t)BB * SS * HH * 4);
  float*  qbuf    = (float*)alloc((size_t)BB * HH * 4);
  __bf16* ctx_b   = (__bf16*)alloc((size_t)BB * KK * 2);
  __bf16* ctxh_b  = (__bf16*)alloc((size_t)BB * HH * 2);
  float*  gi      = (float*)alloc((size_t)BB * 3 * HH * 4);
  float*  gh      = (float*)alloc((size_t)BB * 3 * HH * 4);
  float*  h_f     = (float*)alloc((size_t)BB * HH * 4);
  __bf16* h_b     = (__bf16*)alloc((size_t)BB * HH * 2);
  (void)ws_size; (void)in_sizes; (void)n_in; (void)out_size;

  auto cvt = [&](const float* s, __bf16* d, int n) {
    cvt_f32_bf16<<<dim3((n + 255) / 256), dim3(256), 0, stream>>>(s, d, n);
  };
  auto gemm = [&](const __bf16* A0, int lda0, const __bf16* W0, int ldw0, int K0,
                  const __bf16* A1, int lda1, const __bf16* W1, int ldw1, int K1,
                  const float* bias, float* C, long ldc, __bf16* Cb, long ldcb,
                  int M, int N) {
    int Ntiles = N / 16;
    dim3 grid((Ntiles + 31) / 32, M / 16);
    gemm_bf16_kernel<<<grid, dim3(256), 0, stream>>>(A0, lda0, W0, ldw0, K0,
                                                     A1, lda1, W1, ldw1, K1,
                                                     bias, C, ldc, Cb, ldcb, Ntiles);
  };

  // ---- one-time (per call) precompute ----
  cvt(fc1_w,  fc1w_b,  HH * HH);
  cvt(att_wq, attwq_b, HH * HH);
  cvt(att_wk, attwk_b, HH * KK);
  cvt(fc0_w,  fc0w_b,  HH * KK);
  cvt(gru_wih, wih_b,  3 * HH * 2 * HH);
  cvt(gru_whh, whh_b,  3 * HH * HH);
  cvt(fc2_w,  fc2w_b,  VV * HH);
  cvt(features, feat_b, BB * SS * KK);

  gather_embed<<<dim3((BB * TT * HH + 255) / 256), dim3(256), 0, stream>>>(emb, captions, xg_b);
  init_h<<<dim3((BB * HH + 255) / 256), dim3(256), 0, stream>>>(h_f, h_b);

  // x = emb[captions] @ fc1_w^T + fc1_b  -> bf16 (B*TT, H)
  gemm(xg_b, HH, fc1w_b, HH, HH, nullptr, 0, nullptr, 0, 0,
       fc1_b, nullptr, 0, xw_b, HH, BB * TT, HH);
  // keys_proj = features @ att_wk^T -> f32 (B*S, H)
  gemm(feat_b, KK, attwk_b, KK, KK, nullptr, 0, nullptr, 0, 0,
       nullptr, keysp, HH, nullptr, 0, BB * SS, HH);

  // ---- 15 sequential decoder steps ----
  for (int t = 0; t < TT; ++t) {
    // q = h @ att_wq^T
    gemm(h_b, HH, attwq_b, HH, HH, nullptr, 0, nullptr, 0, 0,
         nullptr, qbuf, HH, nullptr, 0, BB, HH);
    // attention -> context (bf16)
    attention_kernel<<<dim3(BB), dim3(256), 0, stream>>>(qbuf, keysp, att_v, features, ctx_b);
    // context = fc0(context)
    gemm(ctx_b, KK, fc0w_b, KK, KK, nullptr, 0, nullptr, 0, 0,
         fc0_b, nullptr, 0, ctxh_b, HH, BB, HH);
    // gi = x_t @ Wih[:, :H]^T + ctx @ Wih[:, H:]^T + bih   (no concat materialized)
    gemm(xw_b + (size_t)t * HH, TT * HH, wih_b, 2 * HH, HH,
         ctxh_b, HH, wih_b + HH, 2 * HH, HH,
         gru_bih, gi, 3 * HH, nullptr, 0, BB, 3 * HH);
    // gh = h @ Whh^T + bhh
    gemm(h_b, HH, whh_b, HH, HH, nullptr, 0, nullptr, 0, 0,
         gru_bhh, gh, 3 * HH, nullptr, 0, BB, 3 * HH);
    // gate math, h update (f32 + bf16)
    gru_gates_kernel<<<dim3((BB * HH + 255) / 256), dim3(256), 0, stream>>>(gi, gh, h_f, h_b);
    // logits: out[:, t, :] = h_new @ fc2_w^T + fc2_b   (row stride TT*V)
    gemm(h_b, HH, fc2w_b, HH, HH, nullptr, 0, nullptr, 0, 0,
         fc2_b, out + (size_t)t * VV, (long)TT * VV, nullptr, 0, BB, VV);
  }
}
